// ResidualBlock_16990890622991
// MI455X (gfx1250) — matrix operand order repaired and checked by reading, not compiled, not run on hardware
//
#include <hip/hip_runtime.h>
#include <hip/hip_bf16.h>
#include <math.h>

#define NB 32
#define NL 1024
#define ND 128
#define NF 16
#define NS 2
#define DT_RANK 8
#define DCONV 4
#define MTOT (NB * NL)

typedef __attribute__((ext_vector_type(16))) __bf16 v16bf;
typedef __attribute__((ext_vector_type(8)))  float  v8f;
typedef __attribute__((ext_vector_type(4)))  float  f32x4;
typedef __attribute__((ext_vector_type(4)))  unsigned int u32x4;

union BFragU { u32x4 q[2]; v16bf v; };

__device__ __forceinline__ float sigmoidf_(float x) { return 1.0f / (1.0f + __expf(-x)); }
__device__ __forceinline__ float siluf_(float x)    { return x * sigmoidf_(x); }

// A fragment (16x32 bf16) direct from row-major global:
//   lane halves [0..7]  = K = kb..kb+7      (kb = lane>=16 ? 8 : 0)
//   lane halves [8..15] = K = kb+16..kb+23
__device__ __forceinline__ v16bf load_a_frag_g(const __bf16* __restrict__ rowk, int kb) {
  BFragU u;
  u.q[0] = *(const u32x4*)(rowk + kb);
  u.q[1] = *(const u32x4*)(rowk + kb + 16);
  return u.v;
}
// B fragment from fragment-ready global layout: [kblk][nblk][lane][16 halves]
__device__ __forceinline__ v16bf load_b_frag_g(const __bf16* __restrict__ fragbase, int lane) {
  BFragU u;
  const __bf16* p = fragbase + lane * 16;
  u.q[0] = *(const u32x4*)(p);
  u.q[1] = *(const u32x4*)(p + 8);
  return u.v;
}
// fragment-ready flat index for B element (k, n): nblks = N/16
__device__ __forceinline__ size_t bfrag_idx(int k, int n, int nblks) {
  int kblk = k >> 5, kk = k & 31;
  int lane = ((kk & 8) ? 16 : 0) + (n & 15);
  return (((size_t)kblk * nblks + (n >> 4)) * 32 + lane) * 16 + ((kk & 16) ? 8 : 0) + (kk & 7);
}

// ---------------- FiLM from gpt_emb ----------------
__global__ void k_film(const float* __restrict__ gpt, const float* __restrict__ w1,
                       const float* __restrict__ b1, const float* __restrict__ w2,
                       const float* __restrict__ b2, float* __restrict__ gb) {
  int b = blockIdx.x, t = threadIdx.x;  // 128 threads
  __shared__ float g[ND], t1[ND];
  g[t] = gpt[b * ND + t];
  __syncthreads();
  float acc = b1[t];
  for (int d = 0; d < ND; ++d) acc += g[d] * w1[t * ND + d];
  t1[t] = siluf_(acc);
  __syncthreads();
  for (int j = t; j < 2 * ND; j += ND) {
    float a2 = b2[j];
    for (int d = 0; d < ND; ++d) a2 += t1[d] * w2[j * ND + d];
    gb[b * 2 * ND + j] = a2;
  }
}

// ---- weights fp32 -> bf16, stored fragment-ready: B[k][n] = W[n][k] ----
__global__ void k_convw(const float* __restrict__ win, const float* __restrict__ wout,
                        __bf16* __restrict__ win_fr, __bf16* __restrict__ wout_fr) {
  int i = blockIdx.x * blockDim.x + threadIdx.x;
  if (i < 2 * ND * ND) {
    int n = i / ND, k = i % ND;                 // W[n][k]
    win_fr[bfrag_idx(k, n, 2 * ND / 16)] = (__bf16)win[i];
  } else {
    int j = i - 2 * ND * ND;
    if (j < ND * ND) {
      int n = j / ND, k = j % ND;
      wout_fr[bfrag_idx(k, n, ND / 16)] = (__bf16)wout[j];
    }
  }
}

// ------- RMSNorm + cross-attn over 16 groups + FiLM; wave-per-node -------
__global__ void k_prelude(const float* __restrict__ x, const float* __restrict__ fg,
                          const float* __restrict__ norm_w, const float* __restrict__ gb,
                          float* __restrict__ x1_out, __bf16* __restrict__ h_bf) {
  __shared__ float fgs[NF * ND];
  __shared__ float nw[ND];
  int b = blockIdx.y;
  for (int i = threadIdx.x; i < NF * ND; i += 256) fgs[i] = fg[(size_t)b * NF * ND + i];
  for (int i = threadIdx.x; i < ND; i += 256) nw[i] = norm_w[i];
  __syncthreads();
  int wave = threadIdx.x >> 5, lane = threadIdx.x & 31;
  int l = blockIdx.x * 8 + wave;
  size_t row = ((size_t)b * NL + l) * ND;
  f32x4 xv = *(const f32x4*)(x + row + lane * 4);
  float ss = xv[0] * xv[0] + xv[1] * xv[1] + xv[2] * xv[2] + xv[3] * xv[3];
#pragma unroll
  for (int m = 16; m >= 1; m >>= 1) ss += __shfl_xor(ss, m, 32);
  float inv = rsqrtf(ss / (float)ND + 1e-5f);
  float x1v[4];
#pragma unroll
  for (int j = 0; j < 4; ++j) x1v[j] = xv[j] * inv * nw[lane * 4 + j];
  f32x4 sx1 = {x1v[0], x1v[1], x1v[2], x1v[3]};
  *(f32x4*)(x1_out + row + lane * 4) = sx1;
  float sc[NF];
  const float scale = 0.08838834764831845f;  // 1/sqrt(128)
#pragma unroll
  for (int f = 0; f < NF; ++f) {
    float p = 0.f;
#pragma unroll
    for (int j = 0; j < 4; ++j) p += x1v[j] * fgs[f * ND + lane * 4 + j];
#pragma unroll
    for (int m = 16; m >= 1; m >>= 1) p += __shfl_xor(p, m, 32);
    sc[f] = p * scale;
  }
  float mx = sc[0];
#pragma unroll
  for (int f = 1; f < NF; ++f) mx = fmaxf(mx, sc[f]);
  float se = 0.f;
#pragma unroll
  for (int f = 0; f < NF; ++f) { sc[f] = __expf(sc[f] - mx); se += sc[f]; }
  float rse = 1.0f / se;
  float hv[4] = {x1v[0], x1v[1], x1v[2], x1v[3]};
#pragma unroll
  for (int f = 0; f < NF; ++f) {
    float a = sc[f] * rse;
#pragma unroll
    for (int j = 0; j < 4; ++j) hv[j] += a * fgs[f * ND + lane * 4 + j];
  }
  __bf16* hp = h_bf + row;
#pragma unroll
  for (int j = 0; j < 4; ++j) {
    int d = lane * 4 + j;
    float gm = gb[b * 2 * ND + d], bt = gb[b * 2 * ND + ND + d];
    hp[d] = (__bf16)(hv[j] * gm + bt);
  }
}

// ---------------- in_proj: [MTOT,128] @ [128,256], LDS-free WMMA ----------------
__global__ void k_inproj(const __bf16* __restrict__ A, const __bf16* __restrict__ Wfr,
                         float* __restrict__ xin, float* __restrict__ resv) {
  int m0 = blockIdx.x * 64;
  int wave = threadIdx.x >> 5, lane = threadIdx.x & 31;
  int msub = wave >> 1, nbase = (wave & 1) * 128;
  int m = m0 + msub * 16 + (lane & 15);
  int kb = (lane & 16) ? 8 : 0;
  v8f acc[8] = {};
#pragma unroll
  for (int k0 = 0; k0 < ND; k0 += 32) {
    v16bf a = load_a_frag_g(A + (size_t)m * ND + k0, kb);
#pragma unroll
    for (int t = 0; t < 8; ++t) {
      int n0 = nbase + t * 16;
      v16bf bf = load_b_frag_g(Wfr + (((size_t)(k0 >> 5) * 16 + (n0 >> 4)) * 32) * 16, lane);
      acc[t] = __builtin_amdgcn_wmma_f32_16x16x32_bf16(false, a, false, bf, (short)0, acc[t], false, false);
    }
  }
  int rl = lane & 15, hi = (lane >> 4) * 8;
#pragma unroll
  for (int t = 0; t < 8; ++t) {
    int n = nbase + t * 16 + rl;
#pragma unroll
    for (int j = 0; j < 8; ++j) {
      int mm = m0 + msub * 16 + hi + j;
      float v = acc[t][j];
      if (n < ND) xin[(size_t)mm * ND + n] = v;
      else        resv[(size_t)mm * ND + (n - ND)] = v;
    }
  }
}

// ------- depthwise causal conv + silu + x_proj + dt_proj + softplus -------
// delta is written fragment-ready (K = l, N = d) for the dis GEMM's B operand.
__global__ void k_ssmparams(const float* __restrict__ xin, const float* __restrict__ cw,
                            const float* __restrict__ cb, const float* __restrict__ xpw,
                            const float* __restrict__ dtw, const float* __restrict__ dtb,
                            float* __restrict__ xs_out, __bf16* __restrict__ delta_fr,
                            float* __restrict__ bmcm) {
  int b = blockIdx.y, l = blockIdx.x, d = threadIdx.x;  // 128 threads
  __shared__ float xs_s[ND];
  __shared__ float xdbl[DT_RANK + 2 * NS];
  float acc = cb[d];
#pragma unroll
  for (int j = 0; j < DCONV; ++j) {
    int ls = l - (DCONV - 1) + j;
    float v = (ls >= 0) ? xin[((size_t)b * NL + ls) * ND + d] : 0.f;
    acc += cw[d * DCONV + j] * v;
  }
  float xsv = siluf_(acc);
  size_t row = ((size_t)b * NL + l) * ND;
  xs_out[row + d] = xsv;
  xs_s[d] = xsv;
  __syncthreads();
  if (d < DT_RANK + 2 * NS) {
    float a = 0.f;
    for (int k = 0; k < ND; ++k) a += xs_s[k] * xpw[d * ND + k];
    xdbl[d] = a;
  }
  __syncthreads();
  float dtv = dtb[d];
#pragma unroll
  for (int r = 0; r < DT_RANK; ++r) dtv += xdbl[r] * dtw[d * DT_RANK + r];
  float delta = (dtv > 20.f) ? dtv : log1pf(__expf(dtv));
  delta_fr[(size_t)b * NL * ND + bfrag_idx(l, d, ND / 16)] = (__bf16)delta;
  if (d < 2 * NS) bmcm[((size_t)b * NL + l) * 4 + d] = xdbl[DT_RANK + d];
}

// ------- delta_p = dis[b] @ delta[b]; dis fp32 streamed NT, cvt in VGPRs; LDS-free -------
__global__ void k_deltamix(const float* __restrict__ dis, const __bf16* __restrict__ dfr_all,
                           float* __restrict__ delta_p) {
  int b = blockIdx.y, m0 = blockIdx.x * 128;
  const float*  disb = dis + (size_t)b * NL * NL;
  const __bf16* dfr  = dfr_all + (size_t)b * NL * ND;
  int wave = threadIdx.x >> 5, lane = threadIdx.x & 31;
  int m  = m0 + wave * 16 + (lane & 15);
  int kb = (lane & 16) ? 8 : 0;
  v8f acc[8] = {};
  for (int k0 = 0; k0 < NL; k0 += 32) {
    const float* ar = disb + (size_t)m * NL + k0 + kb;
    f32x4 a0 = __builtin_nontemporal_load((const f32x4*)(ar));
    f32x4 a1 = __builtin_nontemporal_load((const f32x4*)(ar + 4));
    f32x4 a2 = __builtin_nontemporal_load((const f32x4*)(ar + 16));
    f32x4 a3 = __builtin_nontemporal_load((const f32x4*)(ar + 20));
    v16bf a;
#pragma unroll
    for (int e = 0; e < 4; ++e) {
      a[e]      = (__bf16)a0[e];
      a[4 + e]  = (__bf16)a1[e];
      a[8 + e]  = (__bf16)a2[e];
      a[12 + e] = (__bf16)a3[e];
    }
#pragma unroll
    for (int t = 0; t < 8; ++t) {
      v16bf bf = load_b_frag_g(dfr + (((size_t)(k0 >> 5) * 8 + t) * 32) * 16, lane);
      acc[t] = __builtin_amdgcn_wmma_f32_16x16x32_bf16(false, a, false, bf, (short)0, acc[t], false, false);
    }
  }
  int rl = lane & 15, hi = (lane >> 4) * 8;
  float* dp = delta_p + (size_t)b * NL * ND;
#pragma unroll
  for (int t = 0; t < 8; ++t) {
    int n = t * 16 + rl;
#pragma unroll
    for (int j = 0; j < 8; ++j) {
      int mm = m0 + wave * 16 + hi + j;
      dp[(size_t)mm * ND + n] = acc[t][j];
    }
  }
}

// ------- sequential selective scan over l, fused D-skip + silu gate -------
__global__ void k_scan(const float* __restrict__ delta_p, const float* __restrict__ xs,
                       const float* __restrict__ bmcm, const float* __restrict__ A_log,
                       const float* __restrict__ D_skip, const float* __restrict__ resv,
                       __bf16* __restrict__ ygate) {
  int b = blockIdx.x, d = threadIdx.x;  // 128 threads
  float A0 = -__expf(A_log[d * NS + 0]);
  float A1 = -__expf(A_log[d * NS + 1]);
  float Dv = D_skip[d];
  float s0 = 0.f, s1 = 0.f;
  size_t base = (size_t)b * NL * ND + d;
  size_t bc   = (size_t)b * NL * 4;
  for (int l = 0; l < NL; ++l) {
    float dp = delta_p[base + (size_t)l * ND];
    float xv = xs[base + (size_t)l * ND];
    float B0 = bmcm[bc + l * 4 + 0], B1 = bmcm[bc + l * 4 + 1];
    float C0 = bmcm[bc + l * 4 + 2], C1 = bmcm[bc + l * 4 + 3];
    float bu = dp * xv;
    s0 = __expf(dp * A0) * s0 + bu * B0;
    s1 = __expf(dp * A1) * s1 + bu * B1;
    float y = s0 * C0 + s1 * C1 + xv * Dv;
    float r = resv[base + (size_t)l * ND];
    ygate[base + (size_t)l * ND] = (__bf16)(y * siluf_(r));
  }
}

// ------- out_proj, LDS-free WMMA + finite mask + residual x1 -------
__global__ void k_outproj(const __bf16* __restrict__ A, const __bf16* __restrict__ Wfr,
                          const float* __restrict__ x1, float* __restrict__ out) {
  int m0 = blockIdx.x * 128;
  int wave = threadIdx.x >> 5, lane = threadIdx.x & 31;
  int m  = m0 + wave * 16 + (lane & 15);
  int kb = (lane & 16) ? 8 : 0;
  v8f acc[8] = {};
#pragma unroll
  for (int k0 = 0; k0 < ND; k0 += 32) {
    v16bf a = load_a_frag_g(A + (size_t)m * ND + k0, kb);
#pragma unroll
    for (int t = 0; t < 8; ++t) {
      v16bf bf = load_b_frag_g(Wfr + (((size_t)(k0 >> 5) * 8 + t) * 32) * 16, lane);
      acc[t] = __builtin_amdgcn_wmma_f32_16x16x32_bf16(false, a, false, bf, (short)0, acc[t], false, false);
    }
  }
  int rl = lane & 15, hi = (lane >> 4) * 8;
#pragma unroll
  for (int t = 0; t < 8; ++t) {
    int n = t * 16 + rl;
#pragma unroll
    for (int j = 0; j < 8; ++j) {
      int mm = m0 + wave * 16 + hi + j;
      float v = acc[t][j];
      if (!isfinite(v)) v = 0.f;
      out[(size_t)mm * ND + n] = v + x1[(size_t)mm * ND + n];
    }
  }
}

// ---------------------------------------------------------------------------
extern "C" void kernel_launch(void* const* d_in, const int* in_sizes, int n_in,
                              void* d_out, int out_size, void* d_ws, size_t ws_size,
                              hipStream_t stream) {
  const float* x      = (const float*)d_in[0];
  const float* dis    = (const float*)d_in[1];
  const float* gpt    = (const float*)d_in[2];
  const float* fg     = (const float*)d_in[3];
  const float* norm_w = (const float*)d_in[4];
  const float* in_w   = (const float*)d_in[5];
  const float* conv_w = (const float*)d_in[6];
  const float* conv_b = (const float*)d_in[7];
  const float* xp_w   = (const float*)d_in[8];
  const float* dt_w   = (const float*)d_in[9];
  const float* dt_b   = (const float*)d_in[10];
  const float* A_log  = (const float*)d_in[11];
  const float* D_skip = (const float*)d_in[12];
  const float* out_w  = (const float*)d_in[13];
  const float* fw1    = (const float*)d_in[14];
  const float* fb1    = (const float*)d_in[15];
  const float* fw2    = (const float*)d_in[16];
  const float* fb2    = (const float*)d_in[17];
  float* out = (float*)d_out;

  char* w = (char*)d_ws;
  size_t off = 0;
  auto carve = [&](size_t bytes) {
    void* p = w + off;
    off += (bytes + 255) & ~(size_t)255;
    return p;
  };
  float*  gb       = (float*) carve(NB * 2 * ND * sizeof(float));
  float*  x1       = (float*) carve((size_t)MTOT * ND * sizeof(float));
  __bf16* h_bf     = (__bf16*)carve((size_t)MTOT * ND * sizeof(__bf16));
  float*  xin      = (float*) carve((size_t)MTOT * ND * sizeof(float));
  float*  resv     = (float*) carve((size_t)MTOT * ND * sizeof(float));
  float*  xs       = (float*) carve((size_t)MTOT * ND * sizeof(float));
  __bf16* delta_fr = (__bf16*)carve((size_t)MTOT * ND * sizeof(__bf16));
  float*  bmcm     = (float*) carve((size_t)MTOT * 4 * sizeof(float));
  float*  delta_p  = (float*) carve((size_t)MTOT * ND * sizeof(float));
  __bf16* ygate    = (__bf16*)carve((size_t)MTOT * ND * sizeof(__bf16));
  __bf16* win_fr   = (__bf16*)carve((size_t)2 * ND * ND * sizeof(__bf16));
  __bf16* wout_fr  = (__bf16*)carve((size_t)ND * ND * sizeof(__bf16));
  (void)ws_size; (void)n_in; (void)in_sizes; (void)out_size;

  k_convw<<<(3 * ND * ND + 255) / 256, 256, 0, stream>>>(in_w, out_w, win_fr, wout_fr);
  k_film<<<NB, ND, 0, stream>>>(gpt, fw1, fb1, fw2, fb2, gb);
  k_prelude<<<dim3(NL / 8, NB), 256, 0, stream>>>(x, fg, norm_w, gb, x1, h_bf);
  k_inproj<<<MTOT / 64, 256, 0, stream>>>(h_bf, win_fr, xin, resv);
  k_ssmparams<<<dim3(NL, NB), ND, 0, stream>>>(xin, conv_w, conv_b, xp_w, dt_w, dt_b,
                                               xs, delta_fr, bmcm);
  k_deltamix<<<dim3(NL / 128, NB), 256, 0, stream>>>(dis, delta_fr, delta_p);
  k_scan<<<NB, ND, 0, stream>>>(delta_p, xs, bmcm, A_log, D_skip, resv, ygate);
  k_outproj<<<MTOT / 128, 256, 0, stream>>>(ygate, wout_fr, x1, out);
}
